// DeepseekV3MoE_46420006535178
// MI455X (gfx1250) — matrix operand order repaired and checked by reading, not compiled, not run on hardware
//
#include <hip/hip_runtime.h>

// ---------------------------------------------------------------------------
// DeepseekV3 MoE for gfx1250 (MI455X): bf16 WMMA + fp32 accumulate,
// async global->LDS copies (ASYNCcnt) with double-buffered LDS pipelining.
// T=2048, H=2048, E=8 routed experts (dense, weighted by top-2 renormalized
// combine weights exactly like the reference), I=1024, shared expert IS=2048.
// ---------------------------------------------------------------------------

typedef __attribute__((ext_vector_type(16))) __bf16 v16bf;
typedef __attribute__((ext_vector_type(8)))  __bf16 v8bf;
typedef __attribute__((ext_vector_type(8)))  float  v8f;

__device__ __forceinline__ unsigned short f2bf(float f) {
  union { float f; unsigned int u; } v; v.f = f;
  unsigned int u = v.u;
  return (unsigned short)((u + 0x7FFFu + ((u >> 16) & 1u)) >> 16);  // RNE
}

// gfx1250 async copy: per-lane 16B global -> LDS, tracked by ASYNCcnt.
__device__ __forceinline__ void async_ld16(unsigned lds_addr, const void* gaddr) {
  asm volatile("global_load_async_to_lds_b128 %0, %1, off"
               :: "v"(lds_addr), "v"(gaddr) : "memory");
}
__device__ __forceinline__ void wait_async0() {
  asm volatile("s_wait_asynccnt 0x0" ::: "memory");
}
__device__ __forceinline__ unsigned lds_off(const void* p) {
  return (unsigned)(unsigned long long)p;  // low 32 bits of generic ptr = LDS offset
}

#define BM 128
#define BN 64
#define BK 32
#define KPAD 40   // BK + 8 pad; 80B rows: 16B aligned, breaks bank conflicts

// ---------------- gate: softmax over 8 experts, top-2, renormalize ----------
__global__ __launch_bounds__(256) void gate_kernel(
    const float* __restrict__ x, const float* __restrict__ gw,
    float* __restrict__ cw, int T, int H) {
  int t = blockIdx.x * 8 + (threadIdx.x >> 5);
  int lane = threadIdx.x & 31;
  if (t >= T) return;
  const float* xr = x + (size_t)t * H;
  float acc[8];
#pragma unroll
  for (int e = 0; e < 8; ++e) acc[e] = 0.f;
  for (int h = lane; h < H; h += 32) {
    float xv = xr[h];
#pragma unroll
    for (int e = 0; e < 8; ++e) acc[e] += xv * gw[(size_t)e * H + h];
  }
#pragma unroll
  for (int e = 0; e < 8; ++e) {
#pragma unroll
    for (int off = 16; off > 0; off >>= 1)
      acc[e] += __shfl_xor(acc[e], off, 32);
  }
  if (lane == 0) {
    float mx = acc[0];
#pragma unroll
    for (int e = 1; e < 8; ++e) mx = fmaxf(mx, acc[e]);
    float p[8]; float s = 0.f;
#pragma unroll
    for (int e = 0; e < 8; ++e) { p[e] = __expf(acc[e] - mx); s += p[e]; }
#pragma unroll
    for (int e = 0; e < 8; ++e) p[e] /= s;
    int i1 = 0;
#pragma unroll
    for (int e = 1; e < 8; ++e) if (p[e] > p[i1]) i1 = e;
    int i2 = (i1 == 0) ? 1 : 0;
#pragma unroll
    for (int e = 0; e < 8; ++e) if (e != i1 && p[e] > p[i2]) i2 = e;
    float denom = p[i1] + p[i2];
#pragma unroll
    for (int e = 0; e < 8; ++e)
      cw[(size_t)t * 8 + e] = (e == i1) ? p[i1] / denom
                            : (e == i2) ? p[i2] / denom : 0.f;
  }
}

// ---------------- x fp32 -> bf16 --------------------------------------------
__global__ __launch_bounds__(256) void cvt_kernel(
    const float* __restrict__ x, unsigned short* __restrict__ xb, int n4) {
  int i = blockIdx.x * 256 + threadIdx.x;
  if (i < n4) {
    float4 v = ((const float4*)x)[i];
    unsigned long long pk =  (unsigned long long)f2bf(v.x)
                          | ((unsigned long long)f2bf(v.y) << 16)
                          | ((unsigned long long)f2bf(v.z) << 32)
                          | ((unsigned long long)f2bf(v.w) << 48);
    *(unsigned long long*)(xb + (size_t)i * 4) = pk;
  }
}

// ---------------- fused gate+up SwiGLU GEMM (double-buffered, async A) ------
__global__ __launch_bounds__(256) void gateup_kernel(
    const unsigned short* __restrict__ xb,   // [T][H] bf16
    const float* __restrict__ wgp,           // [ne][H][If] fp32
    const float* __restrict__ wup,           // [ne][H][If] fp32
    unsigned short* __restrict__ hg,         // [ne][T][If] bf16
    int T, int H, int If) {
  __shared__ __align__(16) unsigned short As[2][BM][KPAD];
  __shared__ __align__(16) unsigned short Bg[2][BN][KPAD];
  __shared__ __align__(16) unsigned short Bu[2][BN][KPAD];

  const int e = blockIdx.z;
  const int nblk = blockIdx.x * BN;
  const int mblk = blockIdx.y * BM;
  const float* wge = wgp + (size_t)e * H * If;
  const float* wue = wup + (size_t)e * H * If;

  const int tid = threadIdx.x;
  const int lane = tid & 31;
  const int wave = tid >> 5;
  const int wm = (wave & 3) * 32;   // 4 waves along M
  const int wn = (wave >> 2) * 32;  // 2 waves along N
  const int l15 = lane & 15;
  const int lh  = lane >> 4;

  // staging geometry
  const int aRow = tid >> 2;            // + 64 for second chunk
  const int aPart = (tid & 3) * 8;      // bf16 elements (16B chunk)
  const int wRow = tid >> 4;            // + 16 for second chunk
  const int wCol = (tid & 15) * 4;      // fp32 columns

  const unsigned short* xa_base = xb + (size_t)mblk * H;

  float4 wgr[2], wur[2];

  auto issueA = [&](int k0, int buf) {
    async_ld16(lds_off(&As[buf][aRow][aPart]),
               xa_base + (size_t)aRow * H + k0 + aPart);
    async_ld16(lds_off(&As[buf][aRow + 64][aPart]),
               xa_base + (size_t)(aRow + 64) * H + k0 + aPart);
  };
  auto loadW = [&](int k0) {
    const float* wgt = wge + (size_t)k0 * If + nblk;
    const float* wut = wue + (size_t)k0 * If + nblk;
#pragma unroll
    for (int i = 0; i < 2; ++i) {
      wgr[i] = *(const float4*)(wgt + (size_t)(wRow + i * 16) * If + wCol);
      wur[i] = *(const float4*)(wut + (size_t)(wRow + i * 16) * If + wCol);
    }
  };
  auto storeW = [&](int buf) {
#pragma unroll
    for (int i = 0; i < 2; ++i) {
      int r = wRow + i * 16;
      Bg[buf][wCol + 0][r] = f2bf(wgr[i].x); Bg[buf][wCol + 1][r] = f2bf(wgr[i].y);
      Bg[buf][wCol + 2][r] = f2bf(wgr[i].z); Bg[buf][wCol + 3][r] = f2bf(wgr[i].w);
      Bu[buf][wCol + 0][r] = f2bf(wur[i].x); Bu[buf][wCol + 1][r] = f2bf(wur[i].y);
      Bu[buf][wCol + 2][r] = f2bf(wur[i].z); Bu[buf][wCol + 3][r] = f2bf(wur[i].w);
    }
  };

  v8f cg[2][2], cu[2][2];
#pragma unroll
  for (int i = 0; i < 2; ++i)
#pragma unroll
    for (int j = 0; j < 2; ++j)
#pragma unroll
      for (int r = 0; r < 8; ++r) { cg[i][j][r] = 0.f; cu[i][j][r] = 0.f; }

  // prologue: stage tile 0 into buffer 0
  issueA(0, 0);
  loadW(0);
  storeW(0);
  wait_async0();
  __syncthreads();

  int p = 0;
  for (int k0 = 0; k0 < H; k0 += BK) {
    const int kn = k0 + BK;
    const bool has = kn < H;
    if (has) {
      issueA(kn, p ^ 1);                 // async: global -> LDS, no VGPR transit
      loadW(kn);                         // regular loads, converted after compute
      __builtin_prefetch(wge + (size_t)(kn + BK) * If + nblk, 0, 1);
      __builtin_prefetch(wue + (size_t)(kn + BK) * If + nblk, 0, 1);
    }

    // ---- compute on buffer p ----
    v16bf af[2];
#pragma unroll
    for (int i = 0; i < 2; ++i) {
      const unsigned short* ab = &As[p][wm + i * 16 + l15][lh * 16];
      v8bf* ap = (v8bf*)&af[i];
      ap[0] = *(const v8bf*)(ab);
      ap[1] = *(const v8bf*)(ab + 8);
    }
    v16bf bgf[2], buf2[2];
#pragma unroll
    for (int j = 0; j < 2; ++j) {
      const unsigned short* gb = &Bg[p][wn + j * 16 + l15][lh * 16];
      const unsigned short* ub = &Bu[p][wn + j * 16 + l15][lh * 16];
      v8bf* gp = (v8bf*)&bgf[j];  gp[0] = *(const v8bf*)gb; gp[1] = *(const v8bf*)(gb + 8);
      v8bf* up = (v8bf*)&buf2[j]; up[0] = *(const v8bf*)ub; up[1] = *(const v8bf*)(ub + 8);
    }
#pragma unroll
    for (int i = 0; i < 2; ++i)
#pragma unroll
      for (int j = 0; j < 2; ++j) {
        cg[i][j] = __builtin_amdgcn_wmma_f32_16x16x32_bf16(
            false, af[i], false, bgf[j], (short)0, cg[i][j], false, false);
        cu[i][j] = __builtin_amdgcn_wmma_f32_16x16x32_bf16(
            false, af[i], false, buf2[j], (short)0, cu[i][j], false, false);
      }

    if (has) storeW(p ^ 1);              // convert next W tile into other buffer
    wait_async0();
    __syncthreads();
    p ^= 1;
  }

  unsigned short* outp = hg + (size_t)e * T * If;
#pragma unroll
  for (int i = 0; i < 2; ++i)
#pragma unroll
    for (int j = 0; j < 2; ++j) {
      int gn = nblk + wn + j * 16 + l15;
#pragma unroll
      for (int r = 0; r < 8; ++r) {
        int gm = mblk + wm + i * 16 + lh * 8 + r;   // C layout: M = r + 8*lanehalf
        float g = cg[i][j][r], u = cu[i][j][r];
        float hval = (g / (1.f + __expf(-g))) * u;  // silu(g)*u
        outp[(size_t)gm * If + gn] = f2bf(hval);
      }
    }
}

// ---------------- down-proj + weighted combine (double-buffered, async A) ---
// routed:  out[t][h]  = sum_e cw[t][e] * (hg[e] @ wd[e])[t][h]   (accumulate=0)
// shared:  out[t][h] +=                 (hg[0] @ wd[0])[t][h]    (cw=null, accumulate=1)
__global__ __launch_bounds__(256) void down_kernel(
    const unsigned short* __restrict__ hg,  // [ne][T][If] bf16
    const float* __restrict__ wdp,          // [ne][If][H] fp32
    const float* __restrict__ cw,           // [T][8] or nullptr
    float* __restrict__ outp,               // [T][H]
    int T, int H, int If, int ne, int accumulate) {
  __shared__ __align__(16) unsigned short As[2][BM][KPAD];
  __shared__ __align__(16) unsigned short Bt[2][BN][KPAD];
  __shared__ float cwl[BM][8];

  const int nblk = blockIdx.x * BN;
  const int mblk = blockIdx.y * BM;
  const int tid = threadIdx.x;
  const int lane = tid & 31;
  const int wave = tid >> 5;
  const int wm = (wave & 3) * 32;
  const int wn = (wave >> 2) * 32;
  const int l15 = lane & 15;
  const int lh  = lane >> 4;

  const int aRow = tid >> 2;
  const int aPart = (tid & 3) * 8;
  const int wRow = tid >> 4;
  const int wCol = (tid & 15) * 4;

  if (cw) {
#pragma unroll
    for (int i = 0; i < 4; ++i) {
      int c = tid + i * 256;  // 128*8 entries
      cwl[c >> 3][c & 7] = cw[(size_t)(mblk + (c >> 3)) * 8 + (c & 7)];
    }
  }
  __syncthreads();

  v8f acc[2][2];
#pragma unroll
  for (int i = 0; i < 2; ++i)
#pragma unroll
    for (int j = 0; j < 2; ++j)
#pragma unroll
      for (int r = 0; r < 8; ++r) acc[i][j][r] = 0.f;

  for (int e = 0; e < ne; ++e) {
    const unsigned short* hge = hg + (size_t)e * T * If;
    const float* wde = wdp + (size_t)e * If * H;

    float4 wr[2];
    auto issueA = [&](int k0, int buf) {
      async_ld16(lds_off(&As[buf][aRow][aPart]),
                 hge + (size_t)(mblk + aRow) * If + k0 + aPart);
      async_ld16(lds_off(&As[buf][aRow + 64][aPart]),
                 hge + (size_t)(mblk + aRow + 64) * If + k0 + aPart);
    };
    auto loadW = [&](int k0) {
      const float* wt = wde + (size_t)k0 * H + nblk;
#pragma unroll
      for (int i = 0; i < 2; ++i)
        wr[i] = *(const float4*)(wt + (size_t)(wRow + i * 16) * H + wCol);
    };
    auto storeW = [&](int buf) {
#pragma unroll
      for (int i = 0; i < 2; ++i) {
        int r = wRow + i * 16;
        Bt[buf][wCol + 0][r] = f2bf(wr[i].x); Bt[buf][wCol + 1][r] = f2bf(wr[i].y);
        Bt[buf][wCol + 2][r] = f2bf(wr[i].z); Bt[buf][wCol + 3][r] = f2bf(wr[i].w);
      }
    };

    v8f ce[2][2];
#pragma unroll
    for (int i = 0; i < 2; ++i)
#pragma unroll
      for (int j = 0; j < 2; ++j)
#pragma unroll
        for (int r = 0; r < 8; ++r) ce[i][j][r] = 0.f;

    issueA(0, 0);
    loadW(0);
    storeW(0);
    wait_async0();
    __syncthreads();

    int p = 0;
    for (int k0 = 0; k0 < If; k0 += BK) {
      const int kn = k0 + BK;
      const bool has = kn < If;
      if (has) {
        issueA(kn, p ^ 1);
        loadW(kn);
        __builtin_prefetch(wde + (size_t)(kn + BK) * H + nblk, 0, 1);
      }

      v16bf af[2];
#pragma unroll
      for (int i = 0; i < 2; ++i) {
        const unsigned short* ab = &As[p][wm + i * 16 + l15][lh * 16];
        v8bf* ap = (v8bf*)&af[i];
        ap[0] = *(const v8bf*)(ab);
        ap[1] = *(const v8bf*)(ab + 8);
      }
      v16bf bf[2];
#pragma unroll
      for (int j = 0; j < 2; ++j) {
        const unsigned short* bb = &Bt[p][wn + j * 16 + l15][lh * 16];
        v8bf* bp = (v8bf*)&bf[j];
        bp[0] = *(const v8bf*)bb; bp[1] = *(const v8bf*)(bb + 8);
      }
#pragma unroll
      for (int i = 0; i < 2; ++i)
#pragma unroll
        for (int j = 0; j < 2; ++j)
          ce[i][j] = __builtin_amdgcn_wmma_f32_16x16x32_bf16(
              false, af[i], false, bf[j], (short)0, ce[i][j], false, false);

      if (has) storeW(p ^ 1);
      wait_async0();
      __syncthreads();
      p ^= 1;
    }

    if (cw) {
#pragma unroll
      for (int i = 0; i < 2; ++i)
#pragma unroll
        for (int j = 0; j < 2; ++j)
#pragma unroll
          for (int r = 0; r < 8; ++r) {
            int ml = wm + i * 16 + lh * 8 + r;
            acc[i][j][r] += cwl[ml][e] * ce[i][j][r];
          }
    } else {
#pragma unroll
      for (int i = 0; i < 2; ++i)
#pragma unroll
        for (int j = 0; j < 2; ++j)
#pragma unroll
          for (int r = 0; r < 8; ++r) acc[i][j][r] += ce[i][j][r];
    }
  }

#pragma unroll
  for (int i = 0; i < 2; ++i)
#pragma unroll
    for (int j = 0; j < 2; ++j) {
      int gn = nblk + wn + j * 16 + l15;
#pragma unroll
      for (int r = 0; r < 8; ++r) {
        int gm = mblk + wm + i * 16 + lh * 8 + r;
        size_t idx = (size_t)gm * H + gn;
        float v = acc[i][j][r];
        if (accumulate) v += outp[idx];
        outp[idx] = v;
      }
    }
}

// ---------------------------------------------------------------------------
extern "C" void kernel_launch(void* const* d_in, const int* in_sizes, int n_in,
                              void* d_out, int out_size, void* d_ws, size_t ws_size,
                              hipStream_t stream) {
  constexpr int T = 2048, H = 2048, E = 8, I = 1024, IS = 2048;

  const float* x      = (const float*)d_in[0];
  const float* gate_w = (const float*)d_in[1];
  const float* wg     = (const float*)d_in[2];
  const float* wu     = (const float*)d_in[3];
  const float* wd     = (const float*)d_in[4];
  const float* sg     = (const float*)d_in[5];
  const float* su     = (const float*)d_in[6];
  const float* sd     = (const float*)d_in[7];
  float* out = (float*)d_out;

  // workspace layout (~48.1 MB total)
  char* ws = (char*)d_ws;
  float*          cwp = (float*)ws;                                         // T*E*4   = 64 KB
  unsigned short* xb  = (unsigned short*)(ws + (1u << 16));                 // T*H*2   = 8 MB
  unsigned short* hgr = (unsigned short*)(ws + (1u << 16) + (8u << 20));    // E*T*I*2 = 32 MB
  unsigned short* hgs = (unsigned short*)(ws + (1u << 16) + (40u << 20));   // T*IS*2  = 8 MB

  gate_kernel<<<T / 8, 256, 0, stream>>>(x, gate_w, cwp, T, H);
  cvt_kernel<<<(T * H / 4 + 255) / 256, 256, 0, stream>>>(x, xb, T * H / 4);

  gateup_kernel<<<dim3(I / BN, T / BM, E), 256, 0, stream>>>(xb, wg, wu, hgr, T, H, I);
  gateup_kernel<<<dim3(IS / BN, T / BM, 1), 256, 0, stream>>>(xb, sg, su, hgs, T, H, IS);

  down_kernel<<<dim3(H / BN, T / BM), 256, 0, stream>>>(hgr, wd, cwp, out, T, H, I, E, 0);
  down_kernel<<<dim3(H / BN, T / BM), 256, 0, stream>>>(hgs, sd, nullptr, out, T, H, IS, 1, 1);
}
